// CausalSelfAttention_72619307041481
// MI455X (gfx1250) — matrix operand order repaired and checked by reading, not compiled, not run on hardware
//
#include <hip/hip_runtime.h>

// ---------------------------------------------------------------------------
// CDNA5 (gfx1250) causal self-attention: bf16 WMMA (16x16x32) f32-accumulate,
// async global->LDS tile movement where no conversion is needed.
// ---------------------------------------------------------------------------

typedef __attribute__((ext_vector_type(16))) __bf16 v16bf;
typedef __attribute__((ext_vector_type(8)))  __bf16 bf16x8;
typedef __attribute__((ext_vector_type(8)))  float  v8f;
typedef __attribute__((ext_vector_type(4)))  int    v4i;

union Frag16 {
  v16bf  v;
  bf16x8 h[2];
};

__device__ __forceinline__ v8f wmma_bf16(const Frag16& a, const Frag16& b, v8f c) {
  // (neg_a, A, neg_b, B, c_mod, C, reuse_a, reuse_b)
  return __builtin_amdgcn_wmma_f32_16x16x32_bf16(false, a.v, false, b.v,
                                                 (short)0, c, false, false);
}

// ---- CDNA5 async global->LDS (ASYNCcnt) with synchronous fallback ---------
#if defined(__AMDGCN__) && __has_builtin(__builtin_amdgcn_global_load_async_to_lds_b128)
#define HAVE_ASYNC_LDS 1
#else
#define HAVE_ASYNC_LDS 0
#endif

#if HAVE_ASYNC_LDS
typedef __attribute__((address_space(1))) v4i* as1_v4i_p;   // printed '__device__ *'
typedef __attribute__((address_space(3))) v4i* as3_v4i_p;   // printed '__shared__ *'
#endif

__device__ __forceinline__ void g2l_b128(const void* g, void* l) {
#if HAVE_ASYNC_LDS
  // Cast via integers: flat global addr == AS1 addr; low 32 bits of a flat
  // LDS addr == AS3 offset on AMDGPU.
  as1_v4i_p gp = (as1_v4i_p)(unsigned long long)(uintptr_t)g;
  as3_v4i_p lp = (as3_v4i_p)(unsigned int)(uintptr_t)l;
  __builtin_amdgcn_global_load_async_to_lds_b128(gp, lp, 0, 0);
#else
  *(bf16x8*)l = *(const bf16x8*)g;
#endif
}

__device__ __forceinline__ void async_wait0() {
#if HAVE_ASYNC_LDS
#if __has_builtin(__builtin_amdgcn_s_wait_asynccnt)
  __builtin_amdgcn_s_wait_asynccnt(0);
#else
  asm volatile("s_wait_asynccnt 0" ::: "memory");
#endif
#endif
}

// Problem constants
#define NB    4
#define SEQ   2048
#define NH    12
#define DH    64
#define DM    768
#define MROWS (NB * SEQ)                       // 8192
#define HEADTEN ((size_t)NB * NH * SEQ * DH)   // 6291456 elems per Q/K/V tensor

// ---------------------------------------------------------------------------
// Tiled GEMM: C[M,N] = A[M,K] * B[K,N] + bias[N]
//  A_BF16=false: A is f32 (converted to bf16 on the fly)
//  EPI=0: scatter bf16 into head-major QKV workspace  (Out = bf16 base)
//  EPI=1: plain f32 output row-major [M,N]
// Block: 256 threads (8 waves), tile 128x64, K-step 32.
// ---------------------------------------------------------------------------
template<bool A_BF16, int EPI>
__global__ __launch_bounds__(256) void gemm_tile_kernel(
    const void* __restrict__ Aptr, const float* __restrict__ Bm,
    const float* __restrict__ bias, void* __restrict__ Outp,
    int M, int N, int K)
{
  constexpr int BM = 128, BN = 64, BK = 32, PAD = 8;
  __shared__ __align__(16) __bf16 lA [BM][BK + PAD];
  __shared__ __align__(16) __bf16 lBt[BN][BK + PAD];   // B stored transposed [n][k]

  const int tid  = threadIdx.x;
  const int wave = tid >> 5;           // wave32
  const int lane = tid & 31;
  const int wm   = wave >> 1;          // 0..3 (M direction)
  const int wn   = wave & 1;           // 0..1 (N direction)
  const int row  = lane & 15;
  const int half = lane >> 4;
  const int m0   = blockIdx.y * BM;
  const int n0   = blockIdx.x * BN;

  const v8f vzero = {0,0,0,0,0,0,0,0};
  v8f c[2][2];
  #pragma unroll
  for (int i = 0; i < 2; ++i)
    #pragma unroll
    for (int j = 0; j < 2; ++j) c[i][j] = vzero;

  for (int k0 = 0; k0 < K; k0 += BK) {
    __syncthreads();
    if (A_BF16) {
      const __bf16* A = (const __bf16*)Aptr;
      #pragma unroll
      for (int it = 0; it < 2; ++it) {
        const int r  = (tid >> 2) + it * 64;
        const int c8 = (tid & 3) * 8;
        // pure bf16 copy -> CDNA5 async global->LDS path
        g2l_b128(A + (size_t)(m0 + r) * K + k0 + c8, &lA[r][c8]);
        if (k0 + BK < K)
          __builtin_prefetch(A + (size_t)(m0 + r) * K + k0 + BK + c8, 0, 0);
      }
    } else {
      const float* A = (const float*)Aptr;
      #pragma unroll
      for (int it = 0; it < 4; ++it) {
        const int r  = (tid >> 3) + it * 32;
        const int c4 = (tid & 7) * 4;
        const float4 f = *(const float4*)(A + (size_t)(m0 + r) * K + k0 + c4);
        lA[r][c4 + 0] = (__bf16)f.x;  lA[r][c4 + 1] = (__bf16)f.y;
        lA[r][c4 + 2] = (__bf16)f.z;  lA[r][c4 + 3] = (__bf16)f.w;
        if (k0 + BK < K)
          __builtin_prefetch(A + (size_t)(m0 + r) * K + k0 + BK + c4, 0, 0);
      }
    }
    #pragma unroll
    for (int it = 0; it < 2; ++it) {
      const int kk = (tid >> 4) + it * 16;
      const int n4 = (tid & 15) * 4;
      const float4 f = *(const float4*)(Bm + (size_t)(k0 + kk) * N + n0 + n4);
      lBt[n4 + 0][kk] = (__bf16)f.x;  lBt[n4 + 1][kk] = (__bf16)f.y;
      lBt[n4 + 2][kk] = (__bf16)f.z;  lBt[n4 + 3][kk] = (__bf16)f.w;
      if (k0 + BK < K)
        __builtin_prefetch(Bm + (size_t)(k0 + BK + kk) * N + n0 + n4, 0, 0);
    }
    if (A_BF16) async_wait0();
    __syncthreads();

    // A fragment: lane row = m%16, K runs [half*8, +8) and [16+half*8, +8)
    Frag16 a[2], b[2];
    #pragma unroll
    for (int i = 0; i < 2; ++i) {
      const int r = wm * 32 + i * 16 + row;
      a[i].h[0] = *(const bf16x8*)&lA[r][half * 8];
      a[i].h[1] = *(const bf16x8*)&lA[r][16 + half * 8];
    }
    // B fragment: lane col = n%16, K runs [half*8, +8) and [16+half*8, +8)
    #pragma unroll
    for (int j = 0; j < 2; ++j) {
      const int n = wn * 32 + j * 16 + row;
      b[j].h[0] = *(const bf16x8*)&lBt[n][half * 8];
      b[j].h[1] = *(const bf16x8*)&lBt[n][16 + half * 8];
    }
    #pragma unroll
    for (int i = 0; i < 2; ++i)
      #pragma unroll
      for (int j = 0; j < 2; ++j)
        c[i][j] = wmma_bf16(a[i], b[j], c[i][j]);
  }

  // Epilogue. C layout: element vv of lane L -> (m = vv + 8*(L/16), n = L%16)
  #pragma unroll
  for (int i = 0; i < 2; ++i) {
    #pragma unroll
    for (int j = 0; j < 2; ++j) {
      const int gn = n0 + wn * 32 + j * 16 + row;
      const float bv = bias[gn];
      #pragma unroll
      for (int vv = 0; vv < 8; ++vv) {
        const int gm = m0 + wm * 32 + i * 16 + vv + 8 * half;
        const float val = c[i][j][vv] + bv;
        if (EPI == 0) {
          __bf16* qkv = (__bf16*)Outp;
          const int which = gn / DM;           // 0=Q,1=K,2=V
          const int col   = gn - which * DM;
          const int head  = col >> 6, dh = col & 63;
          const int bb    = gm >> 11, ss = gm & (SEQ - 1);
          qkv[(size_t)which * HEADTEN +
              (((size_t)(bb * NH + head) * SEQ + ss) << 6) + dh] = (__bf16)val;
        } else {
          ((float*)Outp)[(size_t)gm * N + gn] = val;
        }
      }
    }
  }
}

// ---------------------------------------------------------------------------
// Flash attention: block = (b, h, 64 Q rows), 128 threads = 4 waves,
// each wave owns 16 Q rows. Keys processed 32 at a time.
// ---------------------------------------------------------------------------
__global__ __launch_bounds__(128) void attn_kernel(
    const __bf16* __restrict__ Q, const __bf16* __restrict__ Kt,
    const __bf16* __restrict__ V, __bf16* __restrict__ O)
{
  __shared__ __align__(16) __bf16 lK [32][DH + 8];      // [key][dh]
  __shared__ __align__(16) __bf16 lVt[DH][32 + 8];      // [dh][key] (transposed)
  __shared__ __align__(16) __bf16 lP [4][16][32 + 8];   // per-wave P scratch

  const int tid  = threadIdx.x;
  const int wave = tid >> 5, lane = tid & 31;
  const int row  = lane & 15, half = lane >> 4;
  const int q0   = blockIdx.x * 64;
  const int h    = blockIdx.y;
  const int b    = blockIdx.z;
  const size_t base = (size_t)(b * NH + h) * SEQ * DH;

  // Q A-fragments (Dh=64 -> two K=32 fragments), loaded straight from global
  Frag16 qa[2];
  const int qrow = q0 + wave * 16 + row;
  const __bf16* qb = Q + base + (size_t)qrow * DH;
  qa[0].h[0] = *(const bf16x8*)(qb + half * 8);
  qa[0].h[1] = *(const bf16x8*)(qb + 16 + half * 8);
  qa[1].h[0] = *(const bf16x8*)(qb + 32 + half * 8);
  qa[1].h[1] = *(const bf16x8*)(qb + 48 + half * 8);

  const v8f vzero = {0,0,0,0,0,0,0,0};
  float mrow[8], lrow[8];
  v8f o[4];
  #pragma unroll
  for (int t = 0; t < 4; ++t) o[t] = vzero;
  #pragma unroll
  for (int vv = 0; vv < 8; ++vv) { mrow[vv] = -1e30f; lrow[vv] = 0.0f; }

  const int nkb = (q0 + 64) >> 5;           // causal: keys <= q0+63
  for (int kb = 0; kb < nkb; ++kb) {
    const int kbase = kb * 32;
    __syncthreads();
    #pragma unroll
    for (int it = 0; it < 2; ++it) {
      const int r  = (tid >> 3) + it * 16;  // key row 0..31
      const int c8 = (tid & 7) * 8;         // dh chunk
      // K tile: pure bf16 copy -> CDNA5 async global->LDS
      g2l_b128(Kt + base + (size_t)(kbase + r) * DH + c8, &lK[r][c8]);
      // V tile: needs transpose, stage through VGPRs
      const bf16x8 vd = *(const bf16x8*)(V + base + (size_t)(kbase + r) * DH + c8);
      #pragma unroll
      for (int e = 0; e < 8; ++e) lVt[c8 + e][r] = vd[e];
      if (kb + 1 < nkb) {
        __builtin_prefetch(Kt + base + (size_t)(kbase + 32 + r) * DH + c8, 0, 0);
        __builtin_prefetch(V  + base + (size_t)(kbase + 32 + r) * DH + c8, 0, 0);
      }
    }
    async_wait0();
    __syncthreads();

    // scores = Q(16x64) @ K^T : two 16-key tiles, K-dim (dh) split 32+32
    v8f s[2];
    #pragma unroll
    for (int j = 0; j < 2; ++j) {
      Frag16 bk0, bk1;
      const int kr = j * 16 + row;          // lane column = key
      bk0.h[0] = *(const bf16x8*)&lK[kr][half * 8];
      bk0.h[1] = *(const bf16x8*)&lK[kr][16 + half * 8];
      bk1.h[0] = *(const bf16x8*)&lK[kr][32 + half * 8];
      bk1.h[1] = *(const bf16x8*)&lK[kr][48 + half * 8];
      v8f acc = vzero;
      acc = wmma_bf16(qa[0], bk0, acc);
      acc = wmma_bf16(qa[1], bk1, acc);
      s[j] = acc;
    }

    // scale + causal mask + online softmax (row = vv + 8*half, col = lane%16)
    #pragma unroll
    for (int vv = 0; vv < 8; ++vv) {
      const int qr = q0 + wave * 16 + vv + 8 * half;
      #pragma unroll
      for (int j = 0; j < 2; ++j) {
        const int key = kbase + j * 16 + row;
        float sv = s[j][vv] * 0.125f;       // HEAD_DIM^-0.5
        if (key > qr) sv = -1e30f;
        s[j][vv] = sv;
      }
      float t = fmaxf(s[0][vv], s[1][vv]);
      t = fmaxf(t, __shfl_xor(t, 1, 32));
      t = fmaxf(t, __shfl_xor(t, 2, 32));
      t = fmaxf(t, __shfl_xor(t, 4, 32));
      t = fmaxf(t, __shfl_xor(t, 8, 32));
      const float mnew = fmaxf(mrow[vv], t);
      const float cf = __expf(mrow[vv] - mnew);
      const float p0 = __expf(s[0][vv] - mnew);
      const float p1 = __expf(s[1][vv] - mnew);
      float ps = p0 + p1;
      ps += __shfl_xor(ps, 1, 32);
      ps += __shfl_xor(ps, 2, 32);
      ps += __shfl_xor(ps, 4, 32);
      ps += __shfl_xor(ps, 8, 32);
      lrow[vv] = lrow[vv] * cf + ps;
      mrow[vv] = mnew;
      #pragma unroll
      for (int t4 = 0; t4 < 4; ++t4) o[t4][vv] = o[t4][vv] * cf;
      const int m = vv + 8 * half;
      lP[wave][m][row]      = (__bf16)p0;
      lP[wave][m][16 + row] = (__bf16)p1;
    }

    // P (C layout) -> A-fragment via per-wave LDS round trip (same-wave dscnt)
    Frag16 pa;
    pa.h[0] = *(const bf16x8*)&lP[wave][row][half * 8];
    pa.h[1] = *(const bf16x8*)&lP[wave][row][16 + half * 8];

    // o += P(16x32) @ V(32x64): four dh tiles
    #pragma unroll
    for (int t4 = 0; t4 < 4; ++t4) {
      Frag16 bv;
      const int dr = t4 * 16 + row;         // lane column = dh
      bv.h[0] = *(const bf16x8*)&lVt[dr][half * 8];
      bv.h[1] = *(const bf16x8*)&lVt[dr][16 + half * 8];
      o[t4] = wmma_bf16(pa, bv, o[t4]);
    }
  }

  float inv[8];
  #pragma unroll
  for (int vv = 0; vv < 8; ++vv) inv[vv] = 1.0f / lrow[vv];
  #pragma unroll
  for (int t4 = 0; t4 < 4; ++t4) {
    #pragma unroll
    for (int vv = 0; vv < 8; ++vv) {
      const int m = vv + 8 * half;
      const int sidx = q0 + wave * 16 + m;
      O[(size_t)(b * SEQ + sidx) * DM + h * DH + t4 * 16 + row] =
          (__bf16)(o[t4][vv] * inv[vv]);
    }
  }
}

// ---------------------------------------------------------------------------
extern "C" void kernel_launch(void* const* d_in, const int* in_sizes, int n_in,
                              void* d_out, int out_size, void* d_ws, size_t ws_size,
                              hipStream_t stream) {
  (void)in_sizes; (void)n_in; (void)out_size; (void)ws_size;
  const float* x    = (const float*)d_in[0];
  const float* Wqkv = (const float*)d_in[1];
  const float* bqkv = (const float*)d_in[2];
  const float* Wout = (const float*)d_in[3];
  const float* bout = (const float*)d_in[4];
  float* out = (float*)d_out;

  // Workspace (bf16): Q,K,V head-major [B,H,S,64] then attention output [B,S,768]
  __bf16* qw = (__bf16*)d_ws;
  __bf16* kw = qw + HEADTEN;
  __bf16* vw = kw + HEADTEN;
  __bf16* ow = vw + HEADTEN;

  // 1) QKV projection: [8192,768] x [768,2304] + bias -> scattered bf16 Q/K/V
  dim3 g1(3 * DM / 64, MROWS / 128);
  gemm_tile_kernel<false, 0><<<g1, 256, 0, stream>>>(
      (const void*)x, Wqkv, bqkv, (void*)qw, MROWS, 3 * DM, DM);

  // 2) causal flash attention -> bf16 [B,S,768]
  attn_kernel<<<dim3(SEQ / 64, NH, NB), 128, 0, stream>>>(qw, kw, vw, ow);

  // 3) output projection: [8192,768] x [768,768] + bias -> f32 out
  dim3 g2(DM / 64, MROWS / 128);
  gemm_tile_kernel<true, 1><<<g2, 256, 0, stream>>>(
      (const void*)ow, Wout, bout, (void*)out, MROWS, DM, DM);
}
